// Attention_14869176779215
// MI455X (gfx1250) — matrix operand order repaired and checked by reading, not compiled, not run on hardware
//
#include <hip/hip_runtime.h>

typedef __attribute__((ext_vector_type(16))) __bf16        v16bf;
typedef __attribute__((ext_vector_type(8)))  float         v8f;
typedef __attribute__((ext_vector_type(4)))  unsigned int  u32x4;

#define WMMA_BF16(a, b, c) \
  __builtin_amdgcn_wmma_f32_16x16x32_bf16(false, (a), false, (b), (short)0, (c), false, false)

union FragU {
  v16bf bf;
  u32x4 u[2];
};

__device__ __forceinline__ unsigned short f2bf(float x) {
  unsigned int u = __float_as_uint(x);
  u = (u + 0x7FFFu + ((u >> 16) & 1u)) >> 16;
  return (unsigned short)u;
}

// A fragment (16x32 bf16, M x K): lane L (r = L&15, kh = L>>4):
//   elems 0..7  -> K = kh*8 .. kh*8+7 ; elems 8..15 -> K = 16+kh*8 ..
__device__ __forceinline__ v16bf load_a_frag(const unsigned short* base, int ld, int lane) {
  int r  = lane & 15;
  int kh = lane >> 4;
  const unsigned short* p = base + (size_t)r * ld + kh * 8;
  FragU f;
  f.u[0] = *reinterpret_cast<const u32x4*>(p);
  f.u[1] = *reinterpret_cast<const u32x4*>(p + 16);
  return f.bf;
}

// B fragment (32x16 bf16, K x N) from BT storage (row n holds K contiguous):
//   lane L (n = L&15, kh = L>>4): elems 0..7 -> K = kh*8 .. kh*8+7 ; elems 8..15 -> K = 16+kh*8 ..
__device__ __forceinline__ v16bf load_b_frag(const unsigned short* base, int ld, int lane) {
  int r  = lane & 15;
  int kh = lane >> 4;
  const unsigned short* p = base + (size_t)r * ld + kh * 8;
  FragU f;
  f.u[0] = *reinterpret_cast<const u32x4*>(p);
  f.u[1] = *reinterpret_cast<const u32x4*>(p + 16);
  return f.bf;
}

// ---------------- elementwise / packing kernels ----------------

// packed: each thread converts 2 floats -> one u32 (two bf16)
__global__ void cvt_f32_bf16x2(const float* __restrict__ src, unsigned int* __restrict__ dst, int npairs) {
  int i = blockIdx.x * blockDim.x + threadIdx.x;
  if (i < npairs) {
    unsigned int lo = f2bf(src[2 * i]);
    unsigned int hi = f2bf(src[2 * i + 1]);
    dst[i] = lo | (hi << 16);
  }
}

// Coalesced transpose+convert: dst[c*R + r] = bf16(src[r*C + c]); R,C multiples of 32.
// grid = (C/32, R/32), block = (32, 8)
__global__ __launch_bounds__(256) void tr_cvt(const float* __restrict__ src,
                                              unsigned short* __restrict__ dst, int R, int C) {
  __shared__ unsigned short tile[32][33];
  int tx = threadIdx.x, ty = threadIdx.y;
#pragma unroll
  for (int i = ty; i < 32; i += 8) {
    int r = blockIdx.y * 32 + i, c = blockIdx.x * 32 + tx;
    tile[i][tx] = f2bf(src[(size_t)r * C + c]);
  }
  __syncthreads();
#pragma unroll
  for (int i = ty; i < 32; i += 8) {
    int c = blockIdx.x * 32 + i, r = blockIdx.y * 32 + tx;
    dst[(size_t)c * R + r] = tile[tx][i];
  }
}

// RoPE for Q (scale folded in): Qf (N x 1024 fp32) -> Qb [H][N][64] bf16
__global__ void rope_q(const float* __restrict__ Qf, const float* __restrict__ fc,
                       const float* __restrict__ fs, unsigned short* __restrict__ Qb, float scale) {
  int i = blockIdx.x * blockDim.x + threadIdx.x;  // N*16*32
  if (i >= 2048 * 16 * 32) return;
  int n = i >> 9, rem = i & 511, h = rem >> 5, j = rem & 31;
  float c = fc[n * 32 + j], s = fs[n * 32 + j];
  float qr = Qf[(size_t)n * 1024 + h * 64 + 2 * j];
  float qi = Qf[(size_t)n * 1024 + h * 64 + 2 * j + 1];
  size_t o = ((size_t)h * 2048 + n) * 64 + 2 * j;
  Qb[o]     = f2bf((qr * c - qi * s) * scale);
  Qb[o + 1] = f2bf((qr * s + qi * c) * scale);
}

// RoPE for K: Kf (N x 256 fp32) -> Kb [KV][N][64] bf16
__global__ void rope_k(const float* __restrict__ Kf, const float* __restrict__ fc,
                       const float* __restrict__ fs, unsigned short* __restrict__ Kb) {
  int i = blockIdx.x * blockDim.x + threadIdx.x;  // N*4*32
  if (i >= 2048 * 4 * 32) return;
  int n = i >> 7, rem = i & 127, kv = rem >> 5, j = rem & 31;
  float c = fc[n * 32 + j], s = fs[n * 32 + j];
  float kr = Kf[(size_t)n * 256 + kv * 64 + 2 * j];
  float ki = Kf[(size_t)n * 256 + kv * 64 + 2 * j + 1];
  size_t o = ((size_t)kv * 2048 + n) * 64 + 2 * j;
  Kb[o]     = f2bf(kr * c - ki * s);
  Kb[o + 1] = f2bf(kr * s + ki * c);
}

// V pre-transposed: Vf (N x 256 fp32) -> VbT [KV][64(d)][2048(n)] bf16
__global__ void pack_vt(const float* __restrict__ Vf, unsigned short* __restrict__ VbT) {
  int i = blockIdx.x * blockDim.x + threadIdx.x;  // N*256
  if (i >= 2048 * 256) return;
  int n = i >> 8, rem = i & 255, kv = rem >> 6, d = rem & 63;
  VbT[((size_t)kv * 64 + d) * 2048 + n] = f2bf(Vf[i]);
}

// ---------------- WMMA GEMM: C[M x N] = A[M x K] * BT[N x K]^T ----------------
// block = 256 threads (8 waves); tile M=128 x N=64; B chunk 64x128 double-buffered in LDS.
// Requires K % 128 == 0, M % 128 == 0, N % 64 == 0.
__global__ __launch_bounds__(256) void gemm_bf16(const unsigned short* __restrict__ A,
                                                 const unsigned short* __restrict__ BT,
                                                 float* __restrict__ C,
                                                 int M, int N, int K) {
  __shared__ __align__(16) unsigned short Bs[2][64 * 128];  // 16KB x2

  int wave = threadIdx.x >> 5;
  int lane = threadIdx.x & 31;
  int m0 = blockIdx.y * 128 + wave * 16;
  int n0 = blockIdx.x * 64;
  const unsigned short* Arow = A + (size_t)m0 * K;

  // cooperative B chunk: 64 rows x 128 cols = 1024 x b128 vectors, 4 per thread
  auto gload_b = [&](int kc, u32x4* rg) {
#pragma unroll
    for (int i = 0; i < 4; ++i) {
      int v = threadIdx.x + i * 256;
      int row = v >> 4, cv = v & 15;
      rg[i] = *reinterpret_cast<const u32x4*>(BT + (size_t)(n0 + row) * K + kc + cv * 8);
    }
  };
  auto sstore_b = [&](int buf, const u32x4* rg) {
    u32x4* d = reinterpret_cast<u32x4*>(&Bs[buf][0]);
#pragma unroll
    for (int i = 0; i < 4; ++i) d[threadIdx.x + i * 256] = rg[i];
  };

  u32x4 rg[4];
  gload_b(0, rg);
  sstore_b(0, rg);
  __syncthreads();

  v8f acc[4] = {};
  int nch = K >> 7;  // chunks of 128
  for (int c = 0; c < nch; ++c) {
    int kc = c << 7;
    bool more = (c + 1) < nch;
    if (more) gload_b(kc + 128, rg);  // in-flight during compute

    // batch-load the 4 A fragments for this chunk
    v16bf a[4];
#pragma unroll
    for (int kk = 0; kk < 4; ++kk)
      a[kk] = load_a_frag(Arow + kc + kk * 32, K, lane);

    const unsigned short* Bc = &Bs[c & 1][0];
#pragma unroll
    for (int kk = 0; kk < 4; ++kk)
#pragma unroll
      for (int t = 0; t < 4; ++t) {
        v16bf b = load_b_frag(Bc + t * 16 * 128 + kk * 32, 128, lane);
        acc[t] = WMMA_BF16(a[kk], b, acc[t]);
      }

    if (more) sstore_b((c + 1) & 1, rg);
    __syncthreads();
  }

  int rb = (lane < 16) ? 0 : 8;
  int cn = lane & 15;
#pragma unroll
  for (int t = 0; t < 4; ++t)
#pragma unroll
    for (int r = 0; r < 8; ++r)
      C[(size_t)(m0 + rb + r) * N + n0 + t * 16 + cn] = acc[t][r];
}

// ---------------- flash attention ----------------
// grid = (N/64, H); block = 128 (4 waves). Wave w owns q rows [blk*64+w*16, +16).
// Qb [H][N][64] (pre-scaled), Kb [KV][N][64], VbT [KV][64][2048], Ob [N][H*64]
__global__ __launch_bounds__(128) void attn_fwd(const unsigned short* __restrict__ Qb,
                                                const unsigned short* __restrict__ Kb,
                                                const unsigned short* __restrict__ VbT,
                                                unsigned short* __restrict__ Ob) {
  __shared__ __align__(16) unsigned short Ks[2][64 * 64];   // 8KB x2
  __shared__ __align__(16) unsigned short Vts[2][64 * 64];  // 8KB x2
  __shared__ __align__(16) unsigned short Ps[4][16 * 64];   // 8KB

  int h    = blockIdx.y;
  int kv   = h >> 2;  // REPEATS = 4
  int qrow = blockIdx.x * 64;
  int wave = threadIdx.x >> 5;
  int lane = threadIdx.x & 31;

  const unsigned short* Qh  = Qb + ((size_t)h * 2048 + qrow + wave * 16) * 64;
  const unsigned short* Kh  = Kb + (size_t)kv * 2048 * 64;
  const unsigned short* VhT = VbT + (size_t)kv * 64 * 2048;

  v16bf qf0 = load_a_frag(Qh, 64, lane);
  v16bf qf1 = load_a_frag(Qh + 32, 64, lane);

  // 64x64 tiles: 512 b128 vectors each, 4 per thread
  auto gload_kv = [&](int j, u32x4* kr, u32x4* vr) {
#pragma unroll
    for (int i = 0; i < 4; ++i) {
      int v = threadIdx.x + i * 128;
      int row = v >> 3, cv = v & 7;
      kr[i] = *reinterpret_cast<const u32x4*>(Kh + (size_t)(j + row) * 64 + cv * 8);
      vr[i] = *reinterpret_cast<const u32x4*>(VhT + (size_t)row * 2048 + j + cv * 8);
    }
  };
  auto sstore_kv = [&](int buf, const u32x4* kr, const u32x4* vr) {
    u32x4* dk = reinterpret_cast<u32x4*>(&Ks[buf][0]);
    u32x4* dv = reinterpret_cast<u32x4*>(&Vts[buf][0]);
#pragma unroll
    for (int i = 0; i < 4; ++i) {
      dk[threadIdx.x + i * 128] = kr[i];
      dv[threadIdx.x + i * 128] = vr[i];
    }
  };

  u32x4 kr[4], vr[4];
  gload_kv(0, kr, vr);
  sstore_kv(0, kr, vr);
  __syncthreads();

  float mrow[8], lrow[8];
#pragma unroll
  for (int r = 0; r < 8; ++r) { mrow[r] = -1e30f; lrow[r] = 0.f; }
  v8f o[4] = {};

  for (int j = 0; j < 2048; j += 64) {
    int buf = (j >> 6) & 1;
    bool more = (j + 64) < 2048;
    if (more) gload_kv(j + 64, kr, vr);  // overlap with compute

    const unsigned short* Kst = &Ks[buf][0];
    const unsigned short* Vst = &Vts[buf][0];

    // S = (Q*scale) * K^T
    v8f s[4];
#pragma unroll
    for (int t = 0; t < 4; ++t) {
      v8f z = {};
      v16bf b0 = load_b_frag(Kst + t * 16 * 64, 64, lane);
      v16bf b1 = load_b_frag(Kst + t * 16 * 64 + 32, 64, lane);
      z = WMMA_BF16(qf0, b0, z);
      z = WMMA_BF16(qf1, b1, z);
      s[t] = z;
    }

    // online softmax (rows 0..7 in lanes 0-15, rows 8..15 in lanes 16-31)
#pragma unroll
    for (int r = 0; r < 8; ++r) {
      float mx = fmaxf(fmaxf(s[0][r], s[1][r]), fmaxf(s[2][r], s[3][r]));
      mx = fmaxf(mx, __shfl_xor(mx, 1, 32));
      mx = fmaxf(mx, __shfl_xor(mx, 2, 32));
      mx = fmaxf(mx, __shfl_xor(mx, 4, 32));
      mx = fmaxf(mx, __shfl_xor(mx, 8, 32));
      float mnew  = fmaxf(mrow[r], mx);
      float alpha = __expf(mrow[r] - mnew);
      mrow[r] = mnew;
      float tot = 0.f;
#pragma unroll
      for (int t = 0; t < 4; ++t) {
        float p = __expf(s[t][r] - mnew);
        s[t][r] = p;
        tot += p;
      }
      tot += __shfl_xor(tot, 1, 32);
      tot += __shfl_xor(tot, 2, 32);
      tot += __shfl_xor(tot, 4, 32);
      tot += __shfl_xor(tot, 8, 32);
      lrow[r] = lrow[r] * alpha + tot;
#pragma unroll
      for (int t = 0; t < 4; ++t) o[t][r] *= alpha;
    }

    // stage P (bf16) into per-wave LDS tile, re-fragment as A operand
    unsigned short* P = &Ps[wave][0];
    int cb = lane & 15;
    int rb = (lane < 16) ? 0 : 8;
#pragma unroll
    for (int t = 0; t < 4; ++t)
#pragma unroll
      for (int r = 0; r < 8; ++r)
        P[(rb + r) * 64 + t * 16 + cb] = f2bf(s[t][r]);
    asm volatile("" ::: "memory");  // order per-wave LDS stores before reloads

    v16bf p0 = load_a_frag(P, 64, lane);
    v16bf p1 = load_a_frag(P + 32, 64, lane);
#pragma unroll
    for (int dt = 0; dt < 4; ++dt) {
      v16bf b0 = load_b_frag(Vst + dt * 16 * 64, 64, lane);
      v16bf b1 = load_b_frag(Vst + dt * 16 * 64 + 32, 64, lane);
      o[dt] = WMMA_BF16(p0, b0, o[dt]);
      o[dt] = WMMA_BF16(p1, b1, o[dt]);
    }

    if (more) sstore_kv(buf ^ 1, kr, vr);
    __syncthreads();
  }

  // normalize and write Ob[n][h*64 + d]
  int cn = lane & 15;
  int rb = (lane < 16) ? 0 : 8;
#pragma unroll
  for (int r = 0; r < 8; ++r) {
    float inv = 1.0f / lrow[r];
#pragma unroll
    for (int dt = 0; dt < 4; ++dt)
      Ob[(size_t)(qrow + wave * 16 + rb + r) * 1024 + h * 64 + dt * 16 + cn] =
          f2bf(o[dt][r] * inv);
  }
}

// ---------------- launcher ----------------

extern "C" void kernel_launch(void* const* d_in, const int* in_sizes, int n_in,
                              void* d_out, int out_size, void* d_ws, size_t ws_size,
                              hipStream_t stream) {
  (void)in_sizes; (void)n_in; (void)out_size; (void)ws_size;
  const float* x    = (const float*)d_in[0];
  const float* fcos = (const float*)d_in[1];
  const float* fsin = (const float*)d_in[2];
  const float* Wq   = (const float*)d_in[3];
  const float* Wk   = (const float*)d_in[4];
  const float* Wv   = (const float*)d_in[5];
  const float* Wo   = (const float*)d_in[6];
  float* out = (float*)d_out;

  char* ws = (char*)d_ws;
  size_t off = 0;
  auto alloc = [&](size_t bytes) { char* p = ws + off; off += (bytes + 255) & ~size_t(255); return p; };

  unsigned short* xb  = (unsigned short*)alloc(2048 * 1024 * 2);
  unsigned short* WqT = (unsigned short*)alloc(1024 * 1024 * 2);
  unsigned short* WkT = (unsigned short*)alloc(256 * 1024 * 2);
  unsigned short* WvT = (unsigned short*)alloc(256 * 1024 * 2);
  unsigned short* WoT = (unsigned short*)alloc(1024 * 1024 * 2);
  float*          Qf  = (float*)alloc(2048 * 1024 * 4);
  float*          Kf  = (float*)alloc(2048 * 256 * 4);
  float*          Vf  = (float*)alloc(2048 * 256 * 4);
  unsigned short* Qb  = (unsigned short*)alloc(16 * 2048 * 64 * 2);
  unsigned short* Kb  = (unsigned short*)alloc(4 * 2048 * 64 * 2);
  unsigned short* VbT = (unsigned short*)alloc(4 * 64 * 2048 * 2);
  unsigned short* Ob  = (unsigned short*)alloc(2048 * 1024 * 2);

  // 1) convert x, transpose+convert weights to bf16 "BT" form
  cvt_f32_bf16x2<<<(2048 * 1024 / 2) / 256, 256, 0, stream>>>(x, (unsigned int*)xb, 2048 * 1024 / 2);
  tr_cvt<<<dim3(1024 / 32, 1024 / 32), dim3(32, 8), 0, stream>>>(Wq, WqT, 1024, 1024);
  tr_cvt<<<dim3(256 / 32, 1024 / 32), dim3(32, 8), 0, stream>>>(Wk, WkT, 1024, 256);
  tr_cvt<<<dim3(256 / 32, 1024 / 32), dim3(32, 8), 0, stream>>>(Wv, WvT, 1024, 256);
  tr_cvt<<<dim3(1024 / 32, 1024 / 32), dim3(32, 8), 0, stream>>>(Wo, WoT, 1024, 1024);

  // 2) QKV projections (WMMA)
  gemm_bf16<<<dim3(1024 / 64, 2048 / 128), 256, 0, stream>>>(xb, WqT, Qf, 2048, 1024, 1024);
  gemm_bf16<<<dim3(256 / 64, 2048 / 128), 256, 0, stream>>>(xb, WkT, Kf, 2048, 256, 1024);
  gemm_bf16<<<dim3(256 / 64, 2048 / 128), 256, 0, stream>>>(xb, WvT, Vf, 2048, 256, 1024);

  // 3) RoPE + head-major bf16 repack (scale folded into Q), V pre-transposed
  rope_q<<<(2048 * 16 * 32) / 256, 256, 0, stream>>>(Qf, fcos, fsin, Qb, 0.125f);
  rope_k<<<(2048 * 4 * 32) / 256, 256, 0, stream>>>(Kf, fcos, fsin, Kb);
  pack_vt<<<(2048 * 256) / 256, 256, 0, stream>>>(Vf, VbT);

  // 4) flash attention (WMMA)
  attn_fwd<<<dim3(2048 / 64, 16), 128, 0, stream>>>(Qb, Kb, VbT, Ob);

  // 5) output projection (WMMA) -> fp32 out
  gemm_bf16<<<dim3(1024 / 64, 2048 / 128), 256, 0, stream>>>(Ob, WoT, out, 2048, 1024, 1024);
}
